// GETModel_37331855737113
// MI455X (gfx1250) — matrix operand order repaired and checked by reading, not compiled, not run on hardware
//
#include <hip/hip_runtime.h>
#include <math.h>

typedef __attribute__((ext_vector_type(16))) __bf16 v16bf;
typedef __attribute__((ext_vector_type(8)))  __bf16 v8bf;
typedef __attribute__((ext_vector_type(8)))  float  v8f;

namespace {
constexpr int   cN  = 50000;
constexpr int   cD  = 256;
constexpr int   cH  = 4;
constexpr int   cDZ = 64;
constexpr long  cNE = 1600000;
constexpr long  cNT = 100000;
constexpr int   cP2 = 256;   // H*DZ
constexpr int   cP3 = 1024;  // H*R*DZ
constexpr float kLAM2 = 1.0f, kLAM3 = 0.5f, kLAMM = 1.0f;
constexpr float kBETA2 = 1.0f, kBETA3 = 1.0f, kBETAM = 1.0f;
constexpr float kGCLIP = 1.0f, kSCLIP = 10.0f, kDAMP = 0.9999f, kEPS = 1e-5f;
constexpr unsigned kNegInfKey = 0x007FFFFFu;  // f2key(-inf)
}

// ---------- helpers ----------
__device__ __forceinline__ unsigned f2key(float f) {
  unsigned u = __float_as_uint(f);
  return (u & 0x80000000u) ? ~u : (u | 0x80000000u);
}
__device__ __forceinline__ float key2f(unsigned k) {
  return __uint_as_float((k & 0x80000000u) ? (k & 0x7fffffffu) : ~k);
}
__device__ __forceinline__ float wsum(float v) {
#pragma unroll
  for (int m = 16; m > 0; m >>= 1) v += __shfl_xor(v, m, 32);
  return v;
}
__device__ __forceinline__ v8bf cvt8(float4 a, float4 b) {
  v8bf r;
  r[0] = (__bf16)a.x; r[1] = (__bf16)a.y; r[2] = (__bf16)a.z; r[3] = (__bf16)a.w;
  r[4] = (__bf16)b.x; r[5] = (__bf16)b.y; r[6] = (__bf16)b.z; r[7] = (__bf16)b.w;
  return r;
}

// ---------- fills ----------
__global__ void fill_f32(float* p, float v, long n) {
  long i = blockIdx.x * (long)blockDim.x + threadIdx.x;
  long stride = (long)gridDim.x * blockDim.x;
  for (; i < n; i += stride) p[i] = v;
}
__global__ void fill_u32(unsigned* p, unsigned v, long n) {
  long i = blockIdx.x * (long)blockDim.x + threadIdx.x;
  long stride = (long)gridDim.x * blockDim.x;
  for (; i < n; i += stride) p[i] = v;
}

// ---------- LayerNorm forward ----------
__global__ __launch_bounds__(256) void ln_fwd(const float* __restrict__ X,
                                              const float* __restrict__ gamma,
                                              const float* __restrict__ bias,
                                              float* __restrict__ G,
                                              float* __restrict__ mu_out,
                                              float* __restrict__ rstd_out) {
  int row = blockIdx.x * 8 + (threadIdx.x >> 5);
  int lane = threadIdx.x & 31;
  if (row >= cN) return;
  const float* x = X + (long)row * cD;
  float v[8];
  float s = 0.f;
#pragma unroll
  for (int j = 0; j < 8; ++j) { v[j] = x[lane + j * 32]; s += v[j]; }
  s = wsum(s);
  float mu = s * (1.f / cD);
  float q = 0.f;
#pragma unroll
  for (int j = 0; j < 8; ++j) { float d = v[j] - mu; q += d * d; }
  q = wsum(q);
  float rstd = rsqrtf(q * (1.f / cD) + kEPS);
  float* g = G + (long)row * cD;
#pragma unroll
  for (int j = 0; j < 8; ++j) {
    int i = lane + j * 32;
    g[i] = gamma[i] * (v[j] - mu) * rstd + bias[i];
  }
  if (lane == 0) { mu_out[row] = mu; rstd_out[row] = rstd; }
}

// ---------- WMMA GEMM: C[nrows,P] = A[nrows,Dk] * B ----------
// BPT=1: B stored [P,Dk]  -> C = A*B^T  (forward projections)
// BPT=0: B stored [Dk,P]  -> C = A*B    (backward dG accumulation)
// Block tile 64Mx64N, K-step 32; 8 waves (4x2), each wave: 16x32 via 2 wmma.
template <int BPT>
__global__ __launch_bounds__(256)
void gemm_bf16_wmma(const float* __restrict__ A, const float* __restrict__ B,
                    float* __restrict__ C, int nrows, int P, int Dk, int accum) {
  __shared__ __bf16 As[64][40];  // [m][k], 80B rows: 16B-aligned, bank-spread
  __shared__ __bf16 Bs[64][40];  // [n][k]
  const int tid = threadIdx.x;
  const int lane = tid & 31, wave = tid >> 5;
  const int wm = wave >> 1, wn = wave & 1;
  const int lh = lane >> 4, lr = lane & 15;
  const int row0 = blockIdx.y * 64;
  const int col0 = blockIdx.x * 64;
  v8f acc0 = {0.f, 0.f, 0.f, 0.f, 0.f, 0.f, 0.f, 0.f};
  v8f acc1 = {0.f, 0.f, 0.f, 0.f, 0.f, 0.f, 0.f, 0.f};

  // loop-invariant fill coordinates (guard hoisted out of K loop)
  const int ar = tid >> 2, ac8 = (tid & 3) * 8;
  const bool aval = (row0 + ar) < nrows;
  const float* aptr = A + (long)(row0 + ar) * Dk + ac8;

  for (int k0 = 0; k0 < Dk; k0 += 32) {
    __syncthreads();
    {  // A tile 64x32: one 16B LDS store per thread
      v8bf pa = {};
      if (aval) {
        const float4* p = (const float4*)(aptr + k0);
        pa = cvt8(p[0], p[1]);
      }
      *(v8bf*)&As[ar][ac8] = pa;
    }
    if (BPT) {  // B[P,Dk]: rows contiguous along k -> direct vector copy
      const int bn = tid >> 2, bk8 = (tid & 3) * 8;
      const float4* p = (const float4*)(B + (long)(col0 + bn) * Dk + k0 + bk8);
      *(v8bf*)&Bs[bn][bk8] = cvt8(p[0], p[1]);
    } else {    // B[Dk,P]: contiguous along n -> transpose into LDS
      const int bk = tid >> 3, bn8 = (tid & 7) * 8;
      const float4* p = (const float4*)(B + (long)(k0 + bk) * P + col0 + bn8);
      float4 f0 = p[0], f1 = p[1];
      Bs[bn8 + 0][bk] = (__bf16)f0.x; Bs[bn8 + 1][bk] = (__bf16)f0.y;
      Bs[bn8 + 2][bk] = (__bf16)f0.z; Bs[bn8 + 3][bk] = (__bf16)f0.w;
      Bs[bn8 + 4][bk] = (__bf16)f1.x; Bs[bn8 + 5][bk] = (__bf16)f1.y;
      Bs[bn8 + 6][bk] = (__bf16)f1.z; Bs[bn8 + 7][bk] = (__bf16)f1.w;
    }
    __syncthreads();

    // Fragments per ISA 7.12.2: all reads are contiguous 16B DS loads.
    union { v16bf v; v8bf h[2]; } au, b0, b1;
    const int am = wm * 16 + lr;
    au.h[0] = *(const v8bf*)&As[am][lh * 8];        // K = lh*8 + 0..7
    au.h[1] = *(const v8bf*)&As[am][16 + lh * 8];   // K = 16 + lh*8 + 0..7
    const int bn0 = wn * 32 + lr;
    b0.h[0] = *(const v8bf*)&Bs[bn0][lh * 8];       // K = lh*8 + 0..7
    b0.h[1] = *(const v8bf*)&Bs[bn0][16 + lh * 8];  // K = 16 + lh*8 + 0..7
    b1.h[0] = *(const v8bf*)&Bs[bn0 + 16][lh * 8];
    b1.h[1] = *(const v8bf*)&Bs[bn0 + 16][16 + lh * 8];
    acc0 = __builtin_amdgcn_wmma_f32_16x16x32_bf16(false, au.v, false, b0.v,
                                                   (short)0, acc0, false, false);
    acc1 = __builtin_amdgcn_wmma_f32_16x16x32_bf16(false, au.v, false, b1.v,
                                                   (short)0, acc1, false, false);
  }

  const int gc0 = col0 + wn * 32 + lr;
#pragma unroll
  for (int i = 0; i < 8; ++i) {
    int gr = row0 + wm * 16 + lh * 8 + i;  // C layout: VGPR i -> M = 8*lh + i
    if (gr < nrows) {
      long o0 = (long)gr * P + gc0;
      float v0 = acc0[i], v1 = acc1[i];
      if (accum) { v0 += C[o0]; v1 += C[o0 + 16]; }
      C[o0] = v0;
      C[o0 + 16] = v1;
    }
  }
}

// ---------- pairwise edge scores + segment max ----------
__global__ __launch_bounds__(256) void edge_score2(
    const float* __restrict__ Q2, const float* __restrict__ K2,
    const int* __restrict__ c2, const int* __restrict__ u2,
    float* __restrict__ s2, unsigned* __restrict__ m2u) {
  long t = blockIdx.x * (long)blockDim.x + threadIdx.x;
  if (t >= cNE * cH) return;
  long e = t >> 2; int h = (int)(t & 3);
  const float4* q = (const float4*)(Q2 + (long)c2[e] * cP2 + h * cDZ);
  const float4* k = (const float4*)(K2 + (long)u2[e] * cP2 + h * cDZ);
  float s = 0.f;
#pragma unroll
  for (int i = 0; i < 16; ++i) {
    float4 a = q[i], b = k[i];
    s += a.x * b.x + a.y * b.y + a.z * b.z + a.w * b.w;
  }
  s *= kBETA2;
  s2[t] = s;
  atomicMax(m2u + (long)c2[e] * cH + h, f2key(s));
}

// ---------- z[seg] += exp(s - m[seg]) ----------
__global__ __launch_bounds__(256) void seg_accum(
    const float* __restrict__ s, const int* __restrict__ cidx,
    const unsigned* __restrict__ mkey, float* __restrict__ z, long nE) {
  long t = blockIdx.x * (long)blockDim.x + threadIdx.x;
  if (t >= nE * cH) return;
  long e = t >> 2; int h = (int)(t & 3);
  long seg = (long)cidx[e] * cH + h;
  atomicAdd(z + seg, expf(s[t] - key2f(mkey[seg])));
}

// ---------- lse[i] = z>0 ? m+log(z) : 0 ; E += coef*lse ----------
__global__ __launch_bounds__(256) void seg_lse(
    const unsigned* __restrict__ mkey, const float* __restrict__ z,
    float* __restrict__ lse, float coef, float* __restrict__ Eacc) {
  long i = blockIdx.x * (long)blockDim.x + threadIdx.x;
  if (i >= (long)cN * cH) return;
  float zi = z[i];
  float l = (zi > 0.f) ? key2f(mkey[i]) + logf(fmaxf(zi, 1e-30f)) : 0.f;
  lse[i] = l;
  if (l != 0.f) atomicAdd(Eacc, coef * l);
}

// ---------- edge backward: scatter grads into dQ2/dK2 ----------
__global__ __launch_bounds__(256) void edge_bwd2(
    const float* __restrict__ Q2, const float* __restrict__ K2,
    const int* __restrict__ c2, const int* __restrict__ u2,
    const float* __restrict__ s2, const float* __restrict__ lse2,
    float* __restrict__ dQ2, float* __restrict__ dK2) {
  long t = blockIdx.x * (long)blockDim.x + threadIdx.x;
  if (t >= cNE * cH) return;
  long e = t >> 2; int h = (int)(t & 3);
  int a = c2[e], b = u2[e];
  float w = expf(s2[t] - lse2[(long)a * cH + h]);
  float coef = -kLAM2 * w;
  const float* q = Q2 + (long)a * cP2 + h * cDZ;
  const float* k = K2 + (long)b * cP2 + h * cDZ;
  float* dq = dQ2 + (long)a * cP2 + h * cDZ;
  float* dk = dK2 + (long)b * cP2 + h * cDZ;
#pragma unroll 8
  for (int i = 0; i < cDZ; ++i) {
    atomicAdd(dq + i, coef * k[i]);
    atomicAdd(dk + i, coef * q[i]);
  }
}

// ---------- triple scores ----------
__global__ __launch_bounds__(256) void tri_score3(
    const float* __restrict__ Q3, const float* __restrict__ K3,
    const float* __restrict__ T, const int* __restrict__ c3,
    const int* __restrict__ u3, const int* __restrict__ v3,
    const int* __restrict__ tt, float* __restrict__ s3,
    unsigned* __restrict__ m3u) {
  long t = blockIdx.x * (long)blockDim.x + threadIdx.x;
  if (t >= cNT * cH) return;
  long tr = t >> 2; int h = (int)(t & 3);
  const float4* q = (const float4*)(Q3 + (long)c3[tr] * cP3 + h * 256);
  const float4* ku = (const float4*)(K3 + (long)u3[tr] * cP3 + h * 256);
  const float4* kv = (const float4*)(K3 + (long)v3[tr] * cP3 + h * 256);
  const float4* tm = (const float4*)(T + (long)tt[tr] * cP3 + h * 256);
  float s = 0.f;
#pragma unroll 8
  for (int i = 0; i < 64; ++i) {
    float4 a = q[i], b = ku[i], c = kv[i], d = tm[i];
    s += a.x * b.x * c.x * d.x + a.y * b.y * c.y * d.y +
         a.z * b.z * c.z * d.z + a.w * b.w * c.w * d.w;
  }
  s *= kBETA3;
  s3[t] = s;
  atomicMax(m3u + (long)c3[tr] * cH + h, f2key(s));
}

// ---------- triple backward ----------
__global__ __launch_bounds__(256) void tri_bwd3(
    const float* __restrict__ Q3, const float* __restrict__ K3,
    const float* __restrict__ T, const int* __restrict__ c3,
    const int* __restrict__ u3, const int* __restrict__ v3,
    const int* __restrict__ tt, const float* __restrict__ s3,
    const float* __restrict__ lse3, float* __restrict__ dQ3,
    float* __restrict__ dK3) {
  long t = blockIdx.x * (long)blockDim.x + threadIdx.x;
  if (t >= cNT * cH) return;
  long tr = t >> 2; int h = (int)(t & 3);
  int ic = c3[tr], iu = u3[tr], iv = v3[tr], im = tt[tr];
  float w = expf(s3[t] - lse3[(long)ic * cH + h]);
  float coef = -kLAM3 * w;
  const float* q = Q3 + (long)ic * cP3 + h * 256;
  const float* ku = K3 + (long)iu * cP3 + h * 256;
  const float* kv = K3 + (long)iv * cP3 + h * 256;
  const float* tm = T + (long)im * cP3 + h * 256;
  float* dq = dQ3 + (long)ic * cP3 + h * 256;
  float* du = dK3 + (long)iu * cP3 + h * 256;
  float* dv = dK3 + (long)iv * cP3 + h * 256;
#pragma unroll 4
  for (int i = 0; i < 256; ++i) {
    float a = q[i], b = ku[i], c = kv[i], d = tm[i];
    atomicAdd(dq + i, coef * b * c * d);
    atomicAdd(du + i, coef * a * c * d);
    atomicAdd(dv + i, coef * a * b * d);
  }
}

// ---------- Hopfield memory term ----------
__global__ __launch_bounds__(256) void mem_term(
    const float* __restrict__ Qm, const float* __restrict__ Km2,
    float* __restrict__ dQm, float* __restrict__ Eacc) {
  long t = blockIdx.x * (long)blockDim.x + threadIdx.x;
  if (t >= (long)cN * cH) return;
  long n = t >> 2; int h = (int)(t & 3);
  const float* q = Qm + n * cP2 + h * cDZ;
  float sm[32];
  float mx = -3.4e38f;
#pragma unroll
  for (int k = 0; k < 32; ++k) {
    const float* km = Km2 + (long)k * cP2 + h * cDZ;
    float s = 0.f;
#pragma unroll
    for (int z = 0; z < cDZ; ++z) s += q[z] * km[z];
    s *= kBETAM;
    sm[k] = s;
    mx = fmaxf(mx, s);
  }
  float zs = 0.f;
#pragma unroll
  for (int k = 0; k < 32; ++k) zs += expf(sm[k] - mx);
  float lse = mx + logf(zs);
  atomicAdd(Eacc, -(kLAMM / kBETAM) * lse);
#pragma unroll
  for (int k = 0; k < 32; ++k) sm[k] = expf(sm[k] - lse);  // softmax p_k
  float* dq = dQm + n * cP2 + h * cDZ;
#pragma unroll 4
  for (int z = 0; z < cDZ; ++z) {
    float a = 0.f;
#pragma unroll
    for (int k = 0; k < 32; ++k) a += sm[k] * Km2[(long)k * cP2 + h * cDZ + z];
    dq[z] = -kLAMM * a;
  }
}

// ---------- LayerNorm backward ----------
__global__ __launch_bounds__(256) void ln_bwd(
    const float* __restrict__ X, const float* __restrict__ mu_in,
    const float* __restrict__ rstd_in, const float* __restrict__ gamma,
    const float* __restrict__ dG, float* __restrict__ dX) {
  int row = blockIdx.x * 8 + (threadIdx.x >> 5);
  int lane = threadIdx.x & 31;
  if (row >= cN) return;
  float mu = mu_in[row], rstd = rstd_in[row];
  const float* x = X + (long)row * cD;
  const float* dg = dG + (long)row * cD;
  float dxh[8], xh[8];
  float s1 = 0.f, s2 = 0.f;
#pragma unroll
  for (int j = 0; j < 8; ++j) {
    int i = lane + j * 32;
    xh[j] = (x[i] - mu) * rstd;
    dxh[j] = dg[i] * gamma[i];
    s1 += dxh[j];
    s2 += dxh[j] * xh[j];
  }
  s1 = wsum(s1); s2 = wsum(s2);
  float* dx = dX + (long)row * cD;
#pragma unroll
  for (int j = 0; j < 8; ++j) {
    int i = lane + j * 32;
    dx[i] = rstd * (dxh[j] - (s1 + xh[j] * s2) * (1.f / cD));
  }
}

// ---------- clipped update ----------
__global__ __launch_bounds__(256) void update_x(
    const float* __restrict__ X, const float* __restrict__ gX,
    const float* __restrict__ step, float* __restrict__ out) {
  int row = blockIdx.x * 8 + (threadIdx.x >> 5);
  int lane = threadIdx.x & 31;
  if (row >= cN) return;
  const float* x = X + (long)row * cD;
  const float* g = gX + (long)row * cD;
  float gv[8];
  float gn2 = 0.f;
#pragma unroll
  for (int j = 0; j < 8; ++j) { gv[j] = g[lane + j * 32]; gn2 += gv[j] * gv[j]; }
  gn2 = wsum(gn2);
  float gscale = kGCLIP / fmaxf(sqrtf(gn2), kGCLIP);
  float ss = step[0] * kDAMP * gscale;
  float xn[8];
  float sn2 = 0.f;
#pragma unroll
  for (int j = 0; j < 8; ++j) {
    xn[j] = x[lane + j * 32] - ss * gv[j];
    sn2 += xn[j] * xn[j];
  }
  sn2 = wsum(sn2);
  float sscale = kSCLIP / fmaxf(sqrtf(sn2), kSCLIP);
  float* o = out + (long)row * cD;
#pragma unroll
  for (int j = 0; j < 8; ++j) o[lane + j * 32] = xn[j] * sscale;
}

__global__ void finalize_E(const float* Eacc, float* outE) {
  if (threadIdx.x == 0 && blockIdx.x == 0) outE[0] = Eacc[0];
}

// ==================== launch ====================
extern "C" void kernel_launch(void* const* d_in, const int* in_sizes, int n_in,
                              void* d_out, int out_size, void* d_ws, size_t ws_size,
                              hipStream_t stream) {
  (void)in_sizes; (void)n_in; (void)out_size; (void)ws_size;
  const float* X     = (const float*)d_in[0];
  const int*   c2    = (const int*)d_in[1];
  const int*   u2    = (const int*)d_in[2];
  const int*   c3    = (const int*)d_in[3];
  const int*   u3    = (const int*)d_in[4];
  const int*   v3    = (const int*)d_in[5];
  const int*   tt    = (const int*)d_in[6];
  // d_in[7] = batch (unused)
  const float* step  = (const float*)d_in[8];
  const float* gamma = (const float*)d_in[9];
  const float* bias  = (const float*)d_in[10];
  const float* W_Q2  = (const float*)d_in[11];
  const float* W_K2  = (const float*)d_in[12];
  const float* W_Q3  = (const float*)d_in[13];
  const float* W_K3  = (const float*)d_in[14];
  const float* T_tau = (const float*)d_in[15];
  const float* W_Qm  = (const float*)d_in[16];
  const float* W_Km  = (const float*)d_in[17];
  const float* B_mem = (const float*)d_in[18];
  float* outX = (float*)d_out;
  float* outE = outX + (long)cN * cD;

  char* w = (char*)d_ws;
  size_t off = 0;
  auto alloc = [&](size_t bytes) -> void* {
    void* p = w + off;
    off = (off + bytes + 255) & ~(size_t)255;
    return p;
  };
  float* G    = (float*)alloc((size_t)cN * cD * 4);
  float* muB  = (float*)alloc((size_t)cN * 4);
  float* rsB  = (float*)alloc((size_t)cN * 4);
  float* Q2   = (float*)alloc((size_t)cN * cP2 * 4);
  float* K2   = (float*)alloc((size_t)cN * cP2 * 4);
  float* Qm   = (float*)alloc((size_t)cN * cP2 * 4);
  float* Q3   = (float*)alloc((size_t)cN * cP3 * 4);
  float* K3   = (float*)alloc((size_t)cN * cP3 * 4);
  float* dQ2  = (float*)alloc((size_t)cN * cP2 * 4);
  float* dK2  = (float*)alloc((size_t)cN * cP2 * 4);
  float* dQm  = (float*)alloc((size_t)cN * cP2 * 4);
  float* dQ3  = (float*)alloc((size_t)cN * cP3 * 4);
  float* dK3  = (float*)alloc((size_t)cN * cP3 * 4);
  float* dG   = (float*)alloc((size_t)cN * cD * 4);
  float* dX   = (float*)alloc((size_t)cN * cD * 4);
  float* s2   = (float*)alloc((size_t)cNE * cH * 4);
  float* s3   = (float*)alloc((size_t)cNT * cH * 4);
  unsigned* m2u = (unsigned*)alloc((size_t)cN * cH * 4);
  float* z2   = (float*)alloc((size_t)cN * cH * 4);
  float* lse2 = (float*)alloc((size_t)cN * cH * 4);
  unsigned* m3u = (unsigned*)alloc((size_t)cN * cH * 4);
  float* z3   = (float*)alloc((size_t)cN * cH * 4);
  float* lse3 = (float*)alloc((size_t)cN * cH * 4);
  float* Km2  = (float*)alloc((size_t)32 * cP2 * 4);
  float* Eacc = (float*)alloc(256);

  const int TPB = 256;
  dim3 blk(TPB);
  const long NH = (long)cN * cH;
  const long EH = cNE * cH;
  const long TH = cNT * cH;

  // init scratch
  fill_f32<<<2048, blk, 0, stream>>>(dQ2, 0.f, (long)cN * cP2);
  fill_f32<<<2048, blk, 0, stream>>>(dK2, 0.f, (long)cN * cP2);
  fill_f32<<<4096, blk, 0, stream>>>(dQ3, 0.f, (long)cN * cP3);
  fill_f32<<<4096, blk, 0, stream>>>(dK3, 0.f, (long)cN * cP3);
  fill_f32<<<64, blk, 0, stream>>>(z2, 0.f, NH);
  fill_f32<<<64, blk, 0, stream>>>(z3, 0.f, NH);
  fill_f32<<<1, blk, 0, stream>>>(Eacc, 0.f, 1);
  fill_u32<<<64, blk, 0, stream>>>(m2u, kNegInfKey, NH);
  fill_u32<<<64, blk, 0, stream>>>(m3u, kNegInfKey, NH);

  // forward LN
  ln_fwd<<<(cN + 7) / 8, blk, 0, stream>>>(X, gamma, bias, G, muB, rsB);

  // forward projections (WMMA, C = G * W^T)
  dim3 g2(cP2 / 64, (cN + 63) / 64), g3(cP3 / 64, (cN + 63) / 64), gk(cP2 / 64, 1);
  gemm_bf16_wmma<1><<<g2, blk, 0, stream>>>(G, W_Q2, Q2, cN, cP2, cD, 0);
  gemm_bf16_wmma<1><<<g2, blk, 0, stream>>>(G, W_K2, K2, cN, cP2, cD, 0);
  gemm_bf16_wmma<1><<<g2, blk, 0, stream>>>(G, W_Qm, Qm, cN, cP2, cD, 0);
  gemm_bf16_wmma<1><<<g3, blk, 0, stream>>>(G, W_Q3, Q3, cN, cP3, cD, 0);
  gemm_bf16_wmma<1><<<g3, blk, 0, stream>>>(G, W_K3, K3, cN, cP3, cD, 0);
  gemm_bf16_wmma<1><<<gk, blk, 0, stream>>>(B_mem, W_Km, Km2, 32, cP2, cD, 0);

  // 2nd-order edge energy
  int ebl = (int)((EH + TPB - 1) / TPB);
  edge_score2<<<ebl, blk, 0, stream>>>(Q2, K2, c2, u2, s2, m2u);
  seg_accum<<<ebl, blk, 0, stream>>>(s2, c2, m2u, z2, cNE);
  seg_lse<<<(int)((NH + TPB - 1) / TPB), blk, 0, stream>>>(m2u, z2, lse2, -(kLAM2 / kBETA2), Eacc);
  edge_bwd2<<<ebl, blk, 0, stream>>>(Q2, K2, c2, u2, s2, lse2, dQ2, dK2);

  // 3rd-order motif energy
  int tbl = (int)((TH + TPB - 1) / TPB);
  tri_score3<<<tbl, blk, 0, stream>>>(Q3, K3, T_tau, c3, u3, v3, tt, s3, m3u);
  seg_accum<<<tbl, blk, 0, stream>>>(s3, c3, m3u, z3, cNT);
  seg_lse<<<(int)((NH + TPB - 1) / TPB), blk, 0, stream>>>(m3u, z3, lse3, -(kLAM3 / kBETA3), Eacc);
  tri_bwd3<<<tbl, blk, 0, stream>>>(Q3, K3, T_tau, c3, u3, v3, tt, s3, lse3, dQ3, dK3);

  // memory energy + dQm
  mem_term<<<(int)((NH + TPB - 1) / TPB), blk, 0, stream>>>(Qm, Km2, dQm, Eacc);

  // dG = dQ2*W_Q2 + dK2*W_K2 + dQm*W_Qm + dQ3*W_Q3 + dK3*W_K3  (WMMA, C = A*B)
  dim3 gb(cD / 64, (cN + 63) / 64);
  gemm_bf16_wmma<0><<<gb, blk, 0, stream>>>(dQ2, W_Q2, dG, cN, cD, cP2, 0);
  gemm_bf16_wmma<0><<<gb, blk, 0, stream>>>(dK2, W_K2, dG, cN, cD, cP2, 1);
  gemm_bf16_wmma<0><<<gb, blk, 0, stream>>>(dQm, W_Qm, dG, cN, cD, cP2, 1);
  gemm_bf16_wmma<0><<<gb, blk, 0, stream>>>(dQ3, W_Q3, dG, cN, cD, cP3, 1);
  gemm_bf16_wmma<0><<<gb, blk, 0, stream>>>(dK3, W_K3, dG, cN, cD, cP3, 1);

  // LN backward -> gX, then clipped update
  ln_bwd<<<(cN + 7) / 8, blk, 0, stream>>>(X, muB, rsB, gamma, dG, dX);
  update_x<<<(cN + 7) / 8, blk, 0, stream>>>(X, dX, step, outX);
  finalize_E<<<1, blk, 0, stream>>>(Eacc, outE);
}
